// ImageMoE_56118042689566
// MI455X (gfx1250) — matrix operand order repaired and checked by reading, not compiled, not run on hardware
//
#include <hip/hip_runtime.h>
#include <hip/hip_bf16.h>
#include <math.h>

// ---------------- CDNA5 WMMA types ----------------
typedef __bf16  v16bf  __attribute__((ext_vector_type(16)));
typedef float   v8f    __attribute__((ext_vector_type(8)));
typedef unsigned int u32x4 __attribute__((ext_vector_type(4)));
typedef unsigned int u32x8 __attribute__((ext_vector_type(8)));

static __device__ inline v8f wmma_bf16(v16bf a, v16bf b, v8f c) {
  // v_wmma_f32_16x16x32_bf16 (neg_a, A, neg_b, B, c_mod, C, reuse_a, reuse_b)
  return __builtin_amdgcn_wmma_f32_16x16x32_bf16(false, a, false, b, (short)0, c, false, false);
}

// 16-bit A-fragment K mapping (ISA 7.12.2): slot(half,e) holds k = e + 8*half + (e>=8 ? 8 : 0).
// klin2: kk in [0,64) -> (s = k-step, idx = half*16+e) within As[s][row][48].
static __device__ inline void klin2(int kk, int& s, int& idx) {
  s = kk >> 5;
  int kr = kk & 31, g = kr >> 3;
  idx = ((g & 1) << 4) + (kr - 8 * ((g + 1) >> 1));
}
// hperm: hidden-dim index n -> permuted global position so that an identity byte copy
// into As[s][r][idx] yields a correct A fragment (dense 64 per chunk).
static __device__ inline int hperm(int n) {
  int kk = n & 63, s = kk >> 5, kr = kk & 31, g = kr >> 3;
  int half = g & 1, e = kr - 8 * ((g + 1) >> 1);
  return (n & ~63) + (s << 5) + (half << 4) + e;
}

// ---------------- TDM: 2D f32 tile -> LDS (descriptor per ISA cdna5 08_async_tensor §8.3/8.4) ----
static __device__ inline void tdm_load_2d_f32(const float* gsrc, unsigned lds_off,
                                              unsigned tensor_w, unsigned tensor_h,
                                              unsigned tile_w, unsigned tile_h,
                                              unsigned row_stride /*elems*/) {
  unsigned long long ga = (unsigned long long)(uintptr_t)gsrc;
  u32x4 g0;
  g0[0] = 1u;                                            // count=1 (valid), no gather/restore
  g0[1] = lds_off;                                       // lds_addr (bytes)
  g0[2] = (unsigned)(ga & 0xFFFFFFFFu);                  // global_addr[31:0]
  g0[3] = (unsigned)((ga >> 32) & 0x01FFFFFFu) | (2u << 30);  // global_addr[56:32] | type=2
  u32x8 g1;
  g1[0] = (2u << 16);                                    // data_size=2 (4 bytes)
  g1[1] = (tensor_w & 0xFFFFu) << 16;                    // tensor_dim0[15:0]
  g1[2] = ((tensor_w >> 16) & 0xFFFFu) | ((tensor_h & 0xFFFFu) << 16);
  g1[3] = ((tensor_h >> 16) & 0xFFFFu) | ((tile_w & 0xFFFFu) << 16);  // | tile_dim0
  g1[4] = (tile_h & 0xFFFFu);                            // tile_dim1 (tile_dim2 = 0)
  g1[5] = row_stride;                                    // tensor_dim0_stride[31:0]
  g1[6] = 0u;                                            // stride[47:32] | dim1_stride lo
  g1[7] = 0u;
  asm volatile("tensor_load_to_lds %0, %1" :: "s"(g0), "s"(g1) : "memory");
}

// ---------------- small kernels ----------------

__global__ void zero_ints(int* p, int n) {
  int i = blockIdx.x * 256 + threadIdx.x;
  if (i < n) p[i] = 0;
}

// patchify (b,3,32,32) -> [b*64, 48] @ W_patch[48,1024] + b_patch
__global__ __launch_bounds__(256) void patch_embed(const float* __restrict__ x,
                                                   const float* __restrict__ Wp,
                                                   const float* __restrict__ bp,
                                                   float* __restrict__ h) {
  int tok = blockIdx.x;
  int b = tok >> 6, n = tok & 63;
  int hp = n >> 3, wp = n & 7;
  __shared__ float patch[48];
  int t = threadIdx.x;
  if (t < 48) {
    int c = t / 16, pp = t % 16, pr = pp >> 2, pc = pp & 3;
    patch[t] = x[((size_t)(b * 3 + c) * 32 + (hp * 4 + pr)) * 32 + (wp * 4 + pc)];
  }
  __syncthreads();
  for (int col = t; col < 1024; col += 256) {
    float acc = bp[col];
#pragma unroll
    for (int k = 0; k < 48; ++k) acc += patch[k] * Wp[k * 1024 + col];
    h[(size_t)tok * 1024 + col] = acc;
  }
}

__global__ __launch_bounds__(256) void layernorm(const float* __restrict__ in,
                                                 const float* __restrict__ g,
                                                 const float* __restrict__ bta,
                                                 float* __restrict__ out) {
  int tok = blockIdx.x;
  const float* xr = in + (size_t)tok * 1024;
  __shared__ float red[256];
  int t = threadIdx.x;
  float s = 0.f;
  for (int i = t; i < 1024; i += 256) s += xr[i];
  red[t] = s; __syncthreads();
  for (int o = 128; o > 0; o >>= 1) { if (t < o) red[t] += red[t + o]; __syncthreads(); }
  float mean = red[0] * (1.f / 1024.f);
  __syncthreads();
  float v = 0.f;
  for (int i = t; i < 1024; i += 256) { float d = xr[i] - mean; v += d * d; }
  red[t] = v; __syncthreads();
  for (int o = 128; o > 0; o >>= 1) { if (t < o) red[t] += red[t + o]; __syncthreads(); }
  float inv = rsqrtf(red[0] * (1.f / 1024.f) + 1e-5f);
  __syncthreads();
  for (int i = t; i < 1024; i += 256)
    out[(size_t)tok * 1024 + i] = (xr[i] - mean) * inv * g[i] + bta[i];
}

// ---------------- 32x128-tile bf16 WMMA GEMM (VALU-staged B) ----------------
// C[M,N] = A[M,1024] @ B[1024,N] (+bias)(+addsrc). grid=(M/32, N/128)
__global__ __launch_bounds__(256) void gemm_k1024(const float* __restrict__ A,
                                                  const float* __restrict__ B,
                                                  const float* __restrict__ bias,
                                                  const float* __restrict__ addsrc,
                                                  float* __restrict__ C, int N) {
  __shared__ alignas(16) __bf16 As[2][32][48];
  __shared__ alignas(16) __bf16 Bs[2][128][48];
  int tid = threadIdx.x, lane = tid & 31, wave = tid >> 5;
  int row0 = blockIdx.x * 32;
  int colblk = blockIdx.y * 128;
  int rowb = (wave >> 2) << 4;        // 0 or 16
  int colb = (wave & 3) << 5;         // 0,32,64,96
  int l15 = lane & 15, half16 = (lane >> 4) << 4;
  v8f acc0 = {}, acc1 = {};
  for (int kb = 0; kb < 1024; kb += 64) {
#pragma unroll
    for (int it = 0; it < 2; ++it) {   // stage A: 32 rows x 64 k
      int f = tid + it * 256;
      int r = f >> 4, k4 = (f & 15) << 2;
      float4 v4 = *(const float4*)(A + (size_t)(row0 + r) * 1024 + kb + k4);
      int s_, i_;
      klin2(k4 + 0, s_, i_); As[s_][r][i_] = (__bf16)v4.x;
      klin2(k4 + 1, s_, i_); As[s_][r][i_] = (__bf16)v4.y;
      klin2(k4 + 2, s_, i_); As[s_][r][i_] = (__bf16)v4.z;
      klin2(k4 + 3, s_, i_); As[s_][r][i_] = (__bf16)v4.w;
    }
#pragma unroll
    for (int it = 0; it < 8; ++it) {   // stage B: 64 k x 128 cols (B slot idx = klin2(k))
      int f = tid + it * 256;
      int k = f >> 5, c4 = (f & 31) << 2;
      float4 v4 = *(const float4*)(B + (size_t)(kb + k) * N + colblk + c4);
      int s_, i_; klin2(k, s_, i_);
      Bs[s_][c4 + 0][i_] = (__bf16)v4.x;
      Bs[s_][c4 + 1][i_] = (__bf16)v4.y;
      Bs[s_][c4 + 2][i_] = (__bf16)v4.z;
      Bs[s_][c4 + 3][i_] = (__bf16)v4.w;
    }
    __syncthreads();
    if (kb + 128 < 1024)
      __builtin_prefetch(B + (size_t)(kb + 128) * N + colblk + lane * 4, 0, 1);
#pragma unroll
    for (int s = 0; s < 2; ++s) {
      v16bf fa  = *(const v16bf*)&As[s][rowb + l15][half16];
      v16bf fb0 = *(const v16bf*)&Bs[s][colb + l15][half16];
      v16bf fb1 = *(const v16bf*)&Bs[s][colb + 16 + l15][half16];
      acc0 = wmma_bf16(fa, fb0, acc0);
      acc1 = wmma_bf16(fa, fb1, acc1);
    }
    __syncthreads();
  }
  int n0 = colblk + colb + l15, n1 = n0 + 16;
  int mrow = row0 + rowb + ((lane >> 4) << 3);
#pragma unroll
  for (int r = 0; r < 8; ++r) {
    float v0 = acc0[r], v1 = acc1[r];
    if (bias)   { v0 += bias[n0]; v1 += bias[n1]; }
    if (addsrc) { v0 += addsrc[(size_t)(mrow + r) * N + n0];
                  v1 += addsrc[(size_t)(mrow + r) * N + n1]; }
    C[(size_t)(mrow + r) * N + n0] = v0;
    C[(size_t)(mrow + r) * N + n1] = v1;
  }
}

// attention per (batch, head)
__global__ __launch_bounds__(256) void attention(const float* __restrict__ q,
                                                 const float* __restrict__ k,
                                                 const float* __restrict__ v,
                                                 float* __restrict__ o) {
  int bh = blockIdx.x, b = bh >> 3, hd = bh & 7;
  __shared__ float sc[64][65];
  int t = threadIdx.x;
  const float scale = 0.08838834764831845f;
  for (int i = t; i < 4096; i += 256) {
    int qi = i >> 6, ki = i & 63;
    const float* qp = q + (size_t)(b * 64 + qi) * 1024 + hd * 128;
    const float* kp = k + (size_t)(b * 64 + ki) * 1024 + hd * 128;
    float s = 0.f;
#pragma unroll 8
    for (int d = 0; d < 128; ++d) s += qp[d] * kp[d];
    sc[qi][ki] = s * scale;
  }
  __syncthreads();
  if (t < 64) {
    float mx = -1e30f;
    for (int j = 0; j < 64; ++j) mx = fmaxf(mx, sc[t][j]);
    float sum = 0.f;
    for (int j = 0; j < 64; ++j) { float e = __expf(sc[t][j] - mx); sc[t][j] = e; sum += e; }
    float inv = 1.f / sum;
    for (int j = 0; j < 64; ++j) sc[t][j] *= inv;
  }
  __syncthreads();
  for (int i = t; i < 64 * 128; i += 256) {
    int qi = i >> 7, d = i & 127;
    float s = 0.f;
    for (int j = 0; j < 64; ++j) s += sc[qi][j] * v[(size_t)(b * 64 + j) * 1024 + hd * 128 + d];
    o[(size_t)(b * 64 + qi) * 1024 + hd * 128 + d] = s;
  }
}

__global__ void add_pos(float* __restrict__ h, const float* __restrict__ pos) {
  int tok = blockIdx.x, n = tok & 63;
  for (int i = threadIdx.x; i < 1024; i += 256)
    h[(size_t)tok * 1024 + i] += pos[(size_t)n * 1024 + i];
}

// Noisy top-2 router + scatter into per-expert token lists.
__global__ __launch_bounds__(256) void router(const float* __restrict__ xin,
                                              const float* __restrict__ rW, const float* __restrict__ rb,
                                              const float* __restrict__ nW, const float* __restrict__ nb,
                                              const float* __restrict__ noise,
                                              int* sel_idx, float* sel_gate, int* cnt, int* elist) {
  int tok = blockIdx.x;
  const float* xr = xin + (size_t)tok * 1024;
  __shared__ float red[256];
  __shared__ float vals[20];
  int t = threadIdx.x;
  for (int o = 0; o < 20; ++o) {
    const float* W = (o < 10) ? rW : nW;
    int e = (o < 10) ? o : o - 10;
    float s = 0.f;
    for (int i = t; i < 1024; i += 256) s += xr[i] * W[i * 10 + e];
    red[t] = s; __syncthreads();
    for (int st = 128; st > 0; st >>= 1) { if (t < st) red[t] += red[t + st]; __syncthreads(); }
    if (t == 0) vals[o] = red[0] + ((o < 10) ? rb[e] : nb[e]);
    __syncthreads();
  }
  if (t == 0) {
    float noisy[10];
    for (int e = 0; e < 10; ++e) {
      float nl = vals[10 + e];
      float sp = (nl > 20.f) ? nl : log1pf(__expf(nl));     // softplus
      noisy[e] = vals[e] + noise[(size_t)tok * 10 + e] * sp;
    }
    int i1 = 0;
    for (int e = 1; e < 10; ++e) if (noisy[e] > noisy[i1]) i1 = e;
    int i2 = (i1 == 0) ? 1 : 0;
    for (int e = 0; e < 10; ++e) if (e != i1 && noisy[e] > noisy[i2]) i2 = e;
    float ex = __expf(noisy[i2] - noisy[i1]);
    float g1 = 1.f / (1.f + ex), g2 = ex / (1.f + ex);      // softmax over kept pair
    sel_idx[tok * 2] = i1;  sel_idx[tok * 2 + 1] = i2;
    sel_gate[tok * 2] = g1; sel_gate[tok * 2 + 1] = g2;
    int p1 = atomicAdd(&cnt[i1], 1); elist[i1 * 2048 + p1] = tok;
    int p2 = atomicAdd(&cnt[i2], 1); elist[i2 * 2048 + p2] = tok;
  }
}

__global__ void prefix10(const int* cnt, int* offs, int* nslots) {
  if (threadIdx.x == 0 && blockIdx.x == 0) {
    int off = 0;
    for (int e = 0; e < 10; ++e) { offs[e] = off; off += (cnt[e] + 31) & ~31; }  // 32-aligned
    nslots[0] = off;
  }
}

__global__ void slotmap(const int* cnt, const int* offs, const int* elist,
                        const int* sel_idx, int* sel_slot,
                        int* slot_token, int* slot_expert) {
  int e = blockIdx.x;
  int c = cnt[e], ca = (c + 31) & ~31, base = offs[e];
  for (int pos = threadIdx.x; pos < ca; pos += 256) {
    int slot = base + pos;
    slot_expert[slot] = e;
    if (pos < c) {
      int tk = elist[e * 2048 + pos];
      slot_token[slot] = tk;
      if (sel_idx[tk * 2] == e) sel_slot[tk * 2] = slot; else sel_slot[tk * 2 + 1] = slot;
    } else {
      slot_token[slot] = 0;  // pad row (never read by combine)
    }
  }
}

// hidden[slot, hperm(n)] = relu(X[token] @ W1[e] + b1[e]) bf16; W1 tile via TDM.  grid=(138,32)
__global__ __launch_bounds__(256) void moe_up(const float* __restrict__ X,
                                              const float* __restrict__ W1,
                                              const float* __restrict__ b1,
                                              const int* __restrict__ slot_token,
                                              const int* __restrict__ slot_expert,
                                              const int* __restrict__ nslots,
                                              __bf16* __restrict__ hidden) {
  int slot0 = blockIdx.x * 32;
  if (slot0 >= nslots[0]) return;
  __shared__ alignas(16) __bf16 As[2][32][48];
  __shared__ alignas(16) float  Bsf[2][64][128];   // TDM double buffer (2 x 32KB)
  __shared__ int rows[32];
  int tid = threadIdx.x, lane = tid & 31, wave = tid >> 5;
  if (tid < 32) rows[tid] = slot_token[slot0 + tid];
  int e = slot_expert[slot0];
  const float* W = W1 + (size_t)e * 1024 * 4096;
  int colblk = blockIdx.y * 128;
  int rowb = (wave >> 2) << 4, colb = (wave & 3) << 5;
  int l15 = lane & 15, half16 = (lane >> 4) << 4;
  v8f acc0 = {}, acc1 = {};
  if (wave == 0)
    tdm_load_2d_f32(W + colblk, (unsigned)(uintptr_t)&Bsf[0][0][0],
                    4096u, 1024u, 128u, 64u, 4096u);
  __syncthreads();   // rows[] visible
  for (int c = 0; c < 16; ++c) {
    int kb = c * 64, buf = c & 1;
    bool more = (c + 1 < 16);
    if (wave == 0 && more)
      tdm_load_2d_f32(W + (size_t)(kb + 64) * 4096 + colblk,
                      (unsigned)(uintptr_t)&Bsf[buf ^ 1][0][0],
                      4096u, 1024u, 128u, 64u, 4096u);
#pragma unroll
    for (int it = 0; it < 2; ++it) {   // stage A (cvt to bf16, fragment layout)
      int f = tid + it * 256;
      int r = f >> 4, k4 = (f & 15) << 2;
      float4 v4 = *(const float4*)(X + (size_t)rows[r] * 1024 + kb + k4);
      int s_, i_;
      klin2(k4 + 0, s_, i_); As[s_][r][i_] = (__bf16)v4.x;
      klin2(k4 + 1, s_, i_); As[s_][r][i_] = (__bf16)v4.y;
      klin2(k4 + 2, s_, i_); As[s_][r][i_] = (__bf16)v4.z;
      klin2(k4 + 3, s_, i_); As[s_][r][i_] = (__bf16)v4.w;
    }
    if (wave == 0) {
      if (more) __builtin_amdgcn_s_wait_tensorcnt(1);   // oldest (chunk c) complete
      else      __builtin_amdgcn_s_wait_tensorcnt(0);
    }
    __syncthreads();
#pragma unroll
    for (int s = 0; s < 2; ++s) {
      v16bf fa = *(const v16bf*)&As[s][rowb + l15][half16];
      v16bf fb0, fb1;
#pragma unroll
      for (int e2 = 0; e2 < 16; ++e2) {
        int krow = s * 32 + (half16 >> 1) + (e2 & 7) + ((e2 >> 3) << 4);
        fb0[e2] = (__bf16)Bsf[buf][krow][colb + l15];
        fb1[e2] = (__bf16)Bsf[buf][krow][colb + 16 + l15];
      }
      acc0 = wmma_bf16(fa, fb0, acc0);
      acc1 = wmma_bf16(fa, fb1, acc1);
    }
    __syncthreads();
  }
  int n0 = colblk + colb + l15, n1 = n0 + 16;
  int mrow = slot0 + rowb + ((lane >> 4) << 3);
  float bb0 = b1[e * 4096 + n0], bb1 = b1[e * 4096 + n1];
  int p0 = hperm(n0), p1 = hperm(n1);
#pragma unroll
  for (int r = 0; r < 8; ++r) {
    hidden[(size_t)(mrow + r) * 4096 + p0] = (__bf16)fmaxf(acc0[r] + bb0, 0.f);
    hidden[(size_t)(mrow + r) * 4096 + p1] = (__bf16)fmaxf(acc1[r] + bb1, 0.f);
  }
}

// yslots[slot,1024] = hidden[slot] @ W2[e] + b2[e]; W2 tile via TDM, A tile via async LDS copy.
// grid=(138,8)
__global__ __launch_bounds__(256) void moe_down(const __bf16* __restrict__ hidden,
                                                const float* __restrict__ W2,
                                                const float* __restrict__ b2,
                                                const int* __restrict__ slot_expert,
                                                const int* __restrict__ nslots,
                                                float* __restrict__ yslots) {
  int slot0 = blockIdx.x * 32;
  if (slot0 >= nslots[0]) return;
  __shared__ alignas(16) __bf16 As[2][32][48];
  __shared__ alignas(16) float  Bsf[2][64][128];
  int tid = threadIdx.x, lane = tid & 31, wave = tid >> 5;
  int e = slot_expert[slot0];
  const float* W = W2 + (size_t)e * 4096 * 1024;
  int colblk = blockIdx.y * 128;
  int rowb = (wave >> 2) << 4, colb = (wave & 3) << 5;
  int l15 = lane & 15, half16 = (lane >> 4) << 4;
  // per-thread async A segment: (row, s, half, eseg) -> 16B copy
  int ar = tid >> 3, au = tid & 7;
  int as_ = au >> 2, ah = (au >> 1) & 1, ae = au & 1;
  unsigned lds_a = (unsigned)(uintptr_t)&As[as_][ar][(ah << 4) + (ae << 3)];
  v8f acc0 = {}, acc1 = {};
  if (wave == 0)
    tdm_load_2d_f32(W + colblk, (unsigned)(uintptr_t)&Bsf[0][0][0],
                    1024u, 4096u, 128u, 64u, 1024u);
  for (int c = 0; c < 64; ++c) {
    int kb = c * 64, buf = c & 1;
    bool more = (c + 1 < 64);
    if (wave == 0 && more)
      tdm_load_2d_f32(W + (size_t)(kb + 64) * 1024 + colblk,
                      (unsigned)(uintptr_t)&Bsf[buf ^ 1][0][0],
                      1024u, 4096u, 128u, 64u, 1024u);
    {  // async copy of hidden tile (already fragment-permuted in global memory)
      unsigned long long ga = (unsigned long long)(uintptr_t)
          (hidden + (size_t)(slot0 + ar) * 4096 + kb + (as_ << 5) + (ah << 4) + (ae << 3));
      asm volatile("global_load_async_to_lds_b128 %0, %1, off"
                   :: "v"(lds_a), "v"(ga) : "memory");
    }
    if (wave == 0) {
      if (more) __builtin_amdgcn_s_wait_tensorcnt(1);
      else      __builtin_amdgcn_s_wait_tensorcnt(0);
    }
    asm volatile("s_wait_asynccnt 0x0" ::: "memory");
    __syncthreads();
#pragma unroll
    for (int s = 0; s < 2; ++s) {
      v16bf fa = *(const v16bf*)&As[s][rowb + l15][half16];
      v16bf fb0, fb1;
#pragma unroll
      for (int e2 = 0; e2 < 16; ++e2) {
        int krow = s * 32 + (half16 >> 1) + (e2 & 7) + ((e2 >> 3) << 4);
        fb0[e2] = (__bf16)Bsf[buf][krow][colb + l15];
        fb1[e2] = (__bf16)Bsf[buf][krow][colb + 16 + l15];
      }
      acc0 = wmma_bf16(fa, fb0, acc0);
      acc1 = wmma_bf16(fa, fb1, acc1);
    }
    __syncthreads();
  }
  int n0 = colblk + colb + l15, n1 = n0 + 16;
  int mrow = slot0 + rowb + ((lane >> 4) << 3);
  float bb0 = b2[e * 1024 + n0], bb1 = b2[e * 1024 + n1];
#pragma unroll
  for (int r = 0; r < 8; ++r) {
    yslots[(size_t)(mrow + r) * 1024 + n0] = acc0[r] + bb0;
    yslots[(size_t)(mrow + r) * 1024 + n1] = acc1[r] + bb1;
  }
}

__global__ void combine(const float* __restrict__ yslots, const int* __restrict__ sel_slot,
                        const float* __restrict__ sel_gate, float* __restrict__ out) {
  int tok = blockIdx.x;
  int   s0 = sel_slot[tok * 2], s1 = sel_slot[tok * 2 + 1];
  float g0 = sel_gate[tok * 2], g1 = sel_gate[tok * 2 + 1];
  for (int i = threadIdx.x; i < 1024; i += 256)
    out[(size_t)tok * 1024 + i] =
        g0 * yslots[(size_t)s0 * 1024 + i] + g1 * yslots[(size_t)s1 * 1024 + i];
}

__global__ void mean_feat(const float* __restrict__ second, float* __restrict__ feat) {
  int b = blockIdx.x;
  for (int d = threadIdx.x; d < 1024; d += 256) {
    float s = 0.f;
    for (int n = 0; n < 64; ++n) s += second[(size_t)(b * 64 + n) * 1024 + d];
    feat[(size_t)b * 1024 + d] = s * (1.f / 64.f);
  }
}

__global__ __launch_bounds__(256) void classifier(const float* __restrict__ feat,
                                                  const float* __restrict__ Wc,
                                                  const float* __restrict__ bc,
                                                  float* __restrict__ cls) {
  int b = blockIdx.x;
  __shared__ float red[256];
  int t = threadIdx.x;
  for (int o = 0; o < 10; ++o) {
    float s = 0.f;
    for (int i = t; i < 1024; i += 256) s += feat[(size_t)b * 1024 + i] * Wc[i * 10 + o];
    red[t] = s; __syncthreads();
    for (int st = 128; st > 0; st >>= 1) { if (t < st) red[t] += red[t + st]; __syncthreads(); }
    if (t == 0) cls[b * 10 + o] = red[0] + bc[o];
    __syncthreads();
  }
}

// ---------------- host orchestration ----------------

extern "C" void kernel_launch(void* const* d_in, const int* in_sizes, int n_in,
                              void* d_out, int out_size, void* d_ws, size_t ws_size,
                              hipStream_t stream) {
  (void)in_sizes; (void)n_in; (void)out_size; (void)ws_size;
  const float* x       = (const float*)d_in[0];
  const float* W_patch = (const float*)d_in[1];
  const float* b_patch = (const float*)d_in[2];
  const float* ln1_g = (const float*)d_in[3],  *ln1_b = (const float*)d_in[4];
  const float* ln2_g = (const float*)d_in[5],  *ln2_b = (const float*)d_in[6];
  const float* ln3_g = (const float*)d_in[7],  *ln3_b = (const float*)d_in[8];
  const float* Wq = (const float*)d_in[9],  *Wk = (const float*)d_in[10];
  const float* Wv = (const float*)d_in[11], *Wo = (const float*)d_in[12];
  const float* bo = (const float*)d_in[13];
  const float* pos_emb = (const float*)d_in[14];
  const float* routeW[2] = {(const float*)d_in[15], (const float*)d_in[24]};
  const float* routeB[2] = {(const float*)d_in[16], (const float*)d_in[25]};
  const float* noiseW[2] = {(const float*)d_in[17], (const float*)d_in[26]};
  const float* noiseB[2] = {(const float*)d_in[18], (const float*)d_in[27]};
  const float* eW1[2]    = {(const float*)d_in[19], (const float*)d_in[28]};
  const float* eB1[2]    = {(const float*)d_in[20], (const float*)d_in[29]};
  const float* eW2[2]    = {(const float*)d_in[21], (const float*)d_in[30]};
  const float* eB2[2]    = {(const float*)d_in[22], (const float*)d_in[31]};
  const float* noiseZ[2] = {(const float*)d_in[23], (const float*)d_in[32]};
  const float* Wc = (const float*)d_in[33];
  const float* bc = (const float*)d_in[34];

  float* out   = (float*)d_out;
  float* first = out;                         // [2048,1024]
  float* secnd = out + (size_t)2048 * 1024;   // [2048,1024]
  float* feat  = secnd + (size_t)2048 * 1024; // [32,1024]
  float* cls   = feat + (size_t)32 * 1024;    // [32,10]

  char* p = (char*)d_ws;
  auto carve = [&](size_t bytes) { char* r = p; p += (bytes + 255) & ~(size_t)255; return r; };
  float*  h      = (float*)carve((size_t)2048 * 1024 * 4);
  float*  lnbuf  = (float*)carve((size_t)2048 * 1024 * 4);
  float*  qb     = (float*)carve((size_t)2048 * 1024 * 4);
  float*  kb     = (float*)carve((size_t)2048 * 1024 * 4);
  float*  vb     = (float*)carve((size_t)2048 * 1024 * 4);
  float*  attno  = (float*)carve((size_t)2048 * 1024 * 4);
  __bf16* hidden = (__bf16*)carve((size_t)4416 * 4096 * 2);
  float*  yslots = (float*)carve((size_t)4416 * 1024 * 4);
  int*    cnt        = (int*)carve(16 * 4);
  int*    offs       = (int*)carve(16 * 4);
  int*    nslots     = (int*)carve(16 * 4);
  int*    sel_idx    = (int*)carve(4096 * 4);
  int*    sel_slot   = (int*)carve(4096 * 4);
  float*  sel_gate   = (float*)carve(4096 * 4);
  int*    elist      = (int*)carve(10 * 2048 * 4);
  int*    slot_token = (int*)carve(4416 * 4);
  int*    slot_exp   = (int*)carve(4416 * 4);

  patch_embed<<<2048, 256, 0, stream>>>(x, W_patch, b_patch, h);
  layernorm<<<2048, 256, 0, stream>>>(h, ln1_g, ln1_b, lnbuf);
  dim3 ggemm(64, 8);  // M=2048/32, N=1024/128
  gemm_k1024<<<ggemm, 256, 0, stream>>>(lnbuf, Wq, nullptr, nullptr, qb, 1024);
  gemm_k1024<<<ggemm, 256, 0, stream>>>(lnbuf, Wk, nullptr, nullptr, kb, 1024);
  gemm_k1024<<<ggemm, 256, 0, stream>>>(lnbuf, Wv, nullptr, nullptr, vb, 1024);
  attention<<<256, 256, 0, stream>>>(qb, kb, vb, attno);
  gemm_k1024<<<ggemm, 256, 0, stream>>>(attno, Wo, bo, /*addsrc=*/h, h, 1024);
  add_pos<<<2048, 256, 0, stream>>>(h, pos_emb);

  for (int L = 0; L < 2; ++L) {
    const float* lng = (L == 0) ? ln2_g : ln3_g;
    const float* lnb = (L == 0) ? ln2_b : ln3_b;
    float* dst = (L == 0) ? first : secnd;
    layernorm<<<2048, 256, 0, stream>>>(h, lng, lnb, lnbuf);
    zero_ints<<<1, 256, 0, stream>>>(cnt, 16);
    router<<<2048, 256, 0, stream>>>(lnbuf, routeW[L], routeB[L], noiseW[L], noiseB[L],
                                     noiseZ[L], sel_idx, sel_gate, cnt, elist);
    prefix10<<<1, 32, 0, stream>>>(cnt, offs, nslots);
    slotmap<<<10, 256, 0, stream>>>(cnt, offs, elist, sel_idx, sel_slot, slot_token, slot_exp);
    moe_up<<<dim3(138, 32), 256, 0, stream>>>(lnbuf, eW1[L], eB1[L],
                                              slot_token, slot_exp, nslots, hidden);
    moe_down<<<dim3(138, 8), 256, 0, stream>>>(hidden, eW2[L], eB2[L],
                                               slot_exp, nslots, yslots);
    combine<<<2048, 256, 0, stream>>>(yslots, sel_slot, sel_gate, dst);
  }

  mean_feat<<<32, 256, 0, stream>>>(secnd, feat);
  classifier<<<32, 256, 0, stream>>>(feat, Wc, bc, cls);
}
